// LoRaModel_86938728005645
// MI455X (gfx1250) — matrix operand order repaired and checked by reading, not compiled, not run on hardware
//
#include <hip/hip_runtime.h>
#include <hip/hip_bf16.h>

typedef __attribute__((ext_vector_type(16))) __bf16 v16bf;
typedef __attribute__((ext_vector_type(8)))  float  v8f;

#define N_IMG   32
#define FDIM    128
#define TDIM    512
#define PPI     65536           // pixels per image (F*T)
#define NPIX    2097152         // 32 * 65536
#define OUTLEN  16352           // hop * (T-1)
#define EPS     1e-5f
#define GEMM_BLOCKS 1024        // 32 images * 32 blocks, 2048 px per block

// -------- workspace layout (float offsets) --------
enum : int {
  WS_C1R     = 0,                    // [32][32] layer1 per-image const (real)
  WS_C1I     = 1024,                 // [32][32] (imag)
  WS_SIGPART = 2048,                 // [32][64][5] signal moment partials
  WS_IM      = 12288,                // [32][5] per-image moments
  WS_PQ      = 12464,               // Pr[32] Qr[32] Pi[32] Qi[32] (folded L1+BN1)
  WS_R       = 12592,               // Rr[32][32] Ri[32][32]
  WS_WBIG    = 14640,               // [64][64] real-ified W2 (row=K in, col=N out)
  WS_B2      = 18736,               // [64] bias
  WS_Y2PART  = 18800,               // [1024][128] y2 sum/sumsq partials
  WS_SCALE2  = 149872,              // [64]
  WS_SHIFT2  = 149936,              // [64]
  WS_Y3PART  = 150000,              // [1024][4]
  WS_BN3     = 154096,              // sc3r sh3r sc3i sh3i
  WS_A3PART  = 154100,              // [1024][5]
  WS_BN4     = 159220,              // ar br gr ai bi gi (folded L4+BN4)
  WS_Y3BUF   = 160000,              // [32][65536] float2
  WS_FRAMES  = WS_Y3BUF + NPIX * 2, // [32][512][128] float2
};

// ---------------- layer1 per-image constants ----------------
__global__ __launch_bounds__(1024) void k_const(
    const float2* __restrict__ dly, const float2* __restrict__ stab,
    const int* __restrict__ sid,
    const float* __restrict__ w1r, const float* __restrict__ w1i,
    const float* __restrict__ b1r, const float* __restrict__ b1i,
    float* __restrict__ ws) {
  int t = threadIdx.x;
  int n = t >> 5, o = t & 31;
  float cr = b1r[o], ci = b1i[o];
  int s = sid[n];
  for (int c = 1; c <= 20; ++c) {
    float2 e = (c <= 10) ? dly[n * 10 + (c - 1)] : stab[s * 10 + (c - 11)];
    float wr = w1r[o * 21 + c], wi = w1i[o * 21 + c];
    cr += wr * e.x - wi * e.y;
    ci += wi * e.x + wr * e.y;
  }
  ws[WS_C1R + n * 32 + o] = cr;
  ws[WS_C1I + n * 32 + o] = ci;
}

// ---------------- per-image signal moments ----------------
__global__ __launch_bounds__(256) void k_sigmom(const float2* __restrict__ sig,
                                                float* __restrict__ ws) {
  __shared__ float red[256][5];
  int tid = threadIdx.x;
  int n = blockIdx.x >> 6, chunk = blockIdx.x & 63;
  float m0 = 0, m1 = 0, m2 = 0, m3 = 0, m4 = 0;
  for (int tt = 0; tt < 4; ++tt) {
    float2 sv = sig[n * PPI + chunk * 1024 + tt * 256 + tid];
    m0 += sv.x; m1 += sv.y; m2 += sv.x * sv.x; m3 += sv.y * sv.y; m4 += sv.x * sv.y;
  }
  red[tid][0] = m0; red[tid][1] = m1; red[tid][2] = m2; red[tid][3] = m3; red[tid][4] = m4;
  __syncthreads();
  for (int s = 128; s > 0; s >>= 1) {
    if (tid < s)
      for (int j = 0; j < 5; ++j) red[tid][j] += red[tid + s][j];
    __syncthreads();
  }
  if (tid == 0)
    for (int j = 0; j < 5; ++j)
      ws[WS_SIGPART + (n * 64 + chunk) * 5 + j] = red[0][j];
}

// ------- finalize: analytic BN1 stats, folds, bf16 GEMM B-matrix -------
__global__ __launch_bounds__(256) void k_fin1(
    const float* __restrict__ w1r, const float* __restrict__ w1i,
    const float* __restrict__ g1r, const float* __restrict__ be1r,
    const float* __restrict__ g1i, const float* __restrict__ be1i,
    const float* __restrict__ w2r, const float* __restrict__ w2i,
    const float* __restrict__ b2r, const float* __restrict__ b2i,
    float* __restrict__ ws) {
  __shared__ float G[5];
  int tid = threadIdx.x;
  if (tid < 32) {
    float m[5] = {0, 0, 0, 0, 0};
    for (int c = 0; c < 64; ++c)
      for (int j = 0; j < 5; ++j) m[j] += ws[WS_SIGPART + (tid * 64 + c) * 5 + j];
    for (int j = 0; j < 5; ++j) ws[WS_IM + tid * 5 + j] = m[j];
  }
  __syncthreads();
  if (tid == 0) {
    for (int j = 0; j < 5; ++j) {
      float s = 0;
      for (int nn = 0; nn < 32; ++nn) s += ws[WS_IM + nn * 5 + j];
      G[j] = s;
    }
  }
  __syncthreads();
  const float invNP = 1.f / (float)NPIX;
  if (tid < 64) {
    int o = tid & 31;
    int im = tid >> 5;  // 0 = real channel, 1 = imag channel
    float a = im ? w1i[o * 21] : w1r[o * 21];
    float b = im ? w1r[o * 21] : -w1i[o * 21];
    const float* C = ws + (im ? WS_C1I : WS_C1R);
    float sumC = 0, cross = 0, sumC2 = 0;
    for (int nn = 0; nn < 32; ++nn) {
      float c = C[nn * 32 + o];
      sumC += c;
      sumC2 += c * c;
      cross += c * (a * ws[WS_IM + nn * 5 + 0] + b * ws[WS_IM + nn * 5 + 1]);
    }
    float mean = (a * G[0] + b * G[1] + (float)PPI * sumC) * invNP;
    float sumY2 = a * a * G[2] + b * b * G[3] + 2.f * a * b * G[4] +
                  2.f * cross + (float)PPI * sumC2;
    float var = sumY2 * invNP - mean * mean;
    float rstd = rsqrtf(var + EPS);
    float g  = im ? g1i[o]  : g1r[o];
    float be = im ? be1i[o] : be1r[o];
    float sc = g * rstd;
    ws[WS_PQ + im * 64 + o]      = sc * a;   // P
    ws[WS_PQ + im * 64 + 32 + o] = sc * b;   // Q
    for (int nn = 0; nn < 32; ++nn)
      ws[WS_R + im * 1024 + nn * 32 + o] = sc * (C[nn * 32 + o] - mean) + be;
  }
  // real-ified layer2 weight: Wbig[k_in][n_out]
  for (int idx = tid; idx < 4096; idx += 256) {
    int k = idx >> 6, nn = idx & 63;
    float v;
    if (nn < 32) v = (k < 32) ? w2r[nn * 32 + k] : -w2i[nn * 32 + (k - 32)];
    else {
      int oo = nn - 32;
      v = (k < 32) ? w2i[oo * 32 + k] : w2r[oo * 32 + (k - 32)];
    }
    ws[WS_WBIG + k * 64 + nn] = v;
  }
  if (tid < 64) ws[WS_B2 + tid] = (tid < 32) ? b2r[tid] : b2i[tid - 32];
}

// ---------------- WMMA pass 1: y2 per-channel moments ----------------
__global__ __launch_bounds__(256) void k_gemm_stats(const float2* __restrict__ sig,
                                                    float* __restrict__ ws) {
  __shared__ float sPQ[128];
  __shared__ float sR[64];
  __shared__ float sS[8][32][4];
  __shared__ float sQm[8][32][4];
  const int bid = blockIdx.x;
  const int n = bid >> 5, blk = bid & 31;
  const int tid = threadIdx.x;
  const int w = tid >> 5, lane = tid & 31;
  if (tid < 128) sPQ[tid] = ws[WS_PQ + tid];
  if (tid < 32) {
    sR[tid]      = ws[WS_R + n * 32 + tid];
    sR[tid + 32] = ws[WS_R + 1024 + n * 32 + tid];
  }
  __syncthreads();
  // B tiles: 32x16 bf16, lanes 0-15 hold K=0-7,16-23, lanes 16-31 hold K=8-15,24-31
  v16bf B[2][4];
  {
    const float* wb = ws + WS_WBIG;
    int kbase = ((lane >> 4) & 1) * 16;
    int nl = lane & 15;
    for (int kt = 0; kt < 2; ++kt)
      for (int nt = 0; nt < 4; ++nt)
#pragma unroll
        for (int e = 0; e < 16; ++e)
          B[kt][nt][e] = (__bf16)wb[(kt * 32 + ((e >> 3) << 4) + (kbase >> 1) + (e & 7)) * 64 + nt * 16 + nl];
  }
  float b2v[4];
  for (int nt = 0; nt < 4; ++nt) b2v[nt] = ws[WS_B2 + nt * 16 + (lane & 15)];
  const float *Pr = sPQ, *Qr = sPQ + 32, *Pi = sPQ + 64, *Qi = sPQ + 96;
  const float *Rr = sR, *Ri = sR + 32;
  float accS[4] = {0, 0, 0, 0}, accQ[4] = {0, 0, 0, 0};
  const int kb = ((lane >> 4) & 1) << 3;
  for (int it = 0; it < 16; ++it) {
    int px = blk * 2048 + it * 128 + w * 16 + (lane & 15);
    __builtin_prefetch((const void*)(sig + n * PPI + px + 128), 0, 1);
    float2 s = sig[n * PPI + px];
    v16bf A0, A1;
#pragma unroll
    for (int e = 0; e < 16; ++e) {
      int k = ((e >> 3) << 4) + kb + (e & 7);  // per-ISA 16-bit A layout
      A0[e] = (__bf16)fmaxf(Pr[k] * s.x + Qr[k] * s.y + Rr[k], 0.f);
      A1[e] = (__bf16)fmaxf(Pi[k] * s.x + Qi[k] * s.y + Ri[k], 0.f);
    }
#pragma unroll
    for (int nt = 0; nt < 4; ++nt) {
      v8f acc = {0.f, 0.f, 0.f, 0.f, 0.f, 0.f, 0.f, 0.f};
      acc = __builtin_amdgcn_wmma_f32_16x16x32_bf16(false, A0, false, B[0][nt],
                                                    (short)0, acc, false, false);
      acc = __builtin_amdgcn_wmma_f32_16x16x32_bf16(false, A1, false, B[1][nt],
                                                    (short)0, acc, false, false);
#pragma unroll
      for (int v = 0; v < 8; ++v) {
        float y = acc[v] + b2v[nt];
        accS[nt] += y;
        accQ[nt] += y * y;
      }
    }
  }
  for (int nt = 0; nt < 4; ++nt) { sS[w][lane][nt] = accS[nt]; sQm[w][lane][nt] = accQ[nt]; }
  __syncthreads();
  if (tid < 64) {
    int c = tid, nt = c >> 4, l0 = c & 15;
    float S = 0, Q = 0;
    for (int ww = 0; ww < 8; ++ww) {
      S += sS[ww][l0][nt] + sS[ww][l0 + 16][nt];
      Q += sQm[ww][l0][nt] + sQm[ww][l0 + 16][nt];
    }
    ws[WS_Y2PART + bid * 128 + c] = S;
    ws[WS_Y2PART + bid * 128 + 64 + c] = Q;
  }
}

__global__ __launch_bounds__(64) void k_fin2(
    const float* __restrict__ g2r, const float* __restrict__ be2r,
    const float* __restrict__ g2i, const float* __restrict__ be2i,
    float* __restrict__ ws) {
  int tid = threadIdx.x;
  const float invNP = 1.f / (float)NPIX;
  float S = 0, Q = 0;
  for (int b = 0; b < GEMM_BLOCKS; ++b) {
    S += ws[WS_Y2PART + b * 128 + tid];
    Q += ws[WS_Y2PART + b * 128 + 64 + tid];
  }
  float m = S * invNP;
  float var = Q * invNP - m * m;
  float rstd = rsqrtf(var + EPS);
  float g  = (tid < 32) ? g2r[tid]  : g2i[tid - 32];
  float be = (tid < 32) ? be2r[tid] : be2i[tid - 32];
  ws[WS_SCALE2 + tid] = g * rstd;
  ws[WS_SHIFT2 + tid] = be - g * rstd * m;
}

// ------- WMMA pass 2: y2 -> BN2+ReLU -> layer3 dot -> y3 (+ moments) -------
__global__ __launch_bounds__(256) void k_gemm_y3(
    const float2* __restrict__ sig, float* __restrict__ ws,
    const float* __restrict__ w3r, const float* __restrict__ w3i,
    const float* __restrict__ b3r, const float* __restrict__ b3i) {
  __shared__ float sPQ[128];
  __shared__ float sR[64];
  __shared__ float sSc[64], sSh[64];
  __shared__ float sw3r[32], sw3i[32];
  __shared__ float a2t[8][16][65];
  __shared__ float red[256][4];
  const int bid = blockIdx.x;
  const int n = bid >> 5, blk = bid & 31;
  const int tid = threadIdx.x;
  const int w = tid >> 5, lane = tid & 31;
  if (tid < 128) sPQ[tid] = ws[WS_PQ + tid];
  if (tid < 32) {
    sR[tid]      = ws[WS_R + n * 32 + tid];
    sR[tid + 32] = ws[WS_R + 1024 + n * 32 + tid];
    sw3r[tid] = w3r[tid];
    sw3i[tid] = w3i[tid];
  }
  if (tid < 64) { sSc[tid] = ws[WS_SCALE2 + tid]; sSh[tid] = ws[WS_SHIFT2 + tid]; }
  __syncthreads();
  v16bf B[2][4];
  {
    const float* wb = ws + WS_WBIG;
    int kbase = ((lane >> 4) & 1) * 16;
    int nl = lane & 15;
    for (int kt = 0; kt < 2; ++kt)
      for (int nt = 0; nt < 4; ++nt)
#pragma unroll
        for (int e = 0; e < 16; ++e)
          B[kt][nt][e] = (__bf16)wb[(kt * 32 + ((e >> 3) << 4) + (kbase >> 1) + (e & 7)) * 64 + nt * 16 + nl];
  }
  float b2v[4];
  for (int nt = 0; nt < 4; ++nt) b2v[nt] = ws[WS_B2 + nt * 16 + (lane & 15)];
  const float *Pr = sPQ, *Qr = sPQ + 32, *Pi = sPQ + 64, *Qi = sPQ + 96;
  const float *Rr = sR, *Ri = sR + 32;
  float2* y3buf = (float2*)(ws + WS_Y3BUF);
  const float vb3r = b3r[0], vb3i = b3i[0];
  float m0 = 0, m1 = 0, m2 = 0, m3 = 0;
  const int kb = ((lane >> 4) & 1) << 3;
  const int rbase = (lane >> 4) << 3;
  for (int it = 0; it < 16; ++it) {
    int px = blk * 2048 + it * 128 + w * 16 + (lane & 15);
    __builtin_prefetch((const void*)(sig + n * PPI + px + 128), 0, 1);
    float2 s = sig[n * PPI + px];
    v16bf A0, A1;
#pragma unroll
    for (int e = 0; e < 16; ++e) {
      int k = ((e >> 3) << 4) + kb + (e & 7);
      A0[e] = (__bf16)fmaxf(Pr[k] * s.x + Qr[k] * s.y + Rr[k], 0.f);
      A1[e] = (__bf16)fmaxf(Pi[k] * s.x + Qi[k] * s.y + Ri[k], 0.f);
    }
#pragma unroll
    for (int nt = 0; nt < 4; ++nt) {
      v8f acc = {0.f, 0.f, 0.f, 0.f, 0.f, 0.f, 0.f, 0.f};
      acc = __builtin_amdgcn_wmma_f32_16x16x32_bf16(false, A0, false, B[0][nt],
                                                    (short)0, acc, false, false);
      acc = __builtin_amdgcn_wmma_f32_16x16x32_bf16(false, A1, false, B[1][nt],
                                                    (short)0, acc, false, false);
      int ch = nt * 16 + (lane & 15);
#pragma unroll
      for (int v = 0; v < 8; ++v) {
        float y = acc[v] + b2v[nt];
        a2t[w][rbase + v][ch] = fmaxf(sSc[ch] * y + sSh[ch], 0.f);  // BN2+ReLU
      }
    }
    __syncthreads();
    if (lane < 16) {  // one lane per pixel: layer3 complex dot (32 -> 1)
      float y3r_ = vb3r, y3i_ = vb3i;
#pragma unroll
      for (int c = 0; c < 32; ++c) {
        float ar = a2t[w][lane][c];
        float ai = a2t[w][lane][c + 32];
        y3r_ += sw3r[c] * ar - sw3i[c] * ai;
        y3i_ += sw3i[c] * ar + sw3r[c] * ai;
      }
      y3buf[n * PPI + px] = make_float2(y3r_, y3i_);
      m0 += y3r_; m1 += y3i_; m2 += y3r_ * y3r_; m3 += y3i_ * y3i_;
    }
    __syncthreads();
  }
  red[tid][0] = (lane < 16) ? m0 : 0.f;
  red[tid][1] = (lane < 16) ? m1 : 0.f;
  red[tid][2] = (lane < 16) ? m2 : 0.f;
  red[tid][3] = (lane < 16) ? m3 : 0.f;
  __syncthreads();
  for (int s = 128; s > 0; s >>= 1) {
    if (tid < s)
      for (int j = 0; j < 4; ++j) red[tid][j] += red[tid + s][j];
    __syncthreads();
  }
  if (tid == 0)
    for (int j = 0; j < 4; ++j) ws[WS_Y3PART + bid * 4 + j] = red[0][j];
}

__global__ __launch_bounds__(128) void k_fin3(
    const float* __restrict__ g3r, const float* __restrict__ be3r,
    const float* __restrict__ g3i, const float* __restrict__ be3i,
    float* __restrict__ ws) {
  __shared__ float s4[4];
  int tid = threadIdx.x;
  if (tid < 4) {
    float s = 0;
    for (int b = 0; b < GEMM_BLOCKS; ++b) s += ws[WS_Y3PART + b * 4 + tid];
    s4[tid] = s;
  }
  __syncthreads();
  if (tid == 0) {
    const float invNP = 1.f / (float)NPIX;
    float mr = s4[0] * invNP, mi = s4[1] * invNP;
    float vr = s4[2] * invNP - mr * mr, vi = s4[3] * invNP - mi * mi;
    float scr = g3r[0] * rsqrtf(vr + EPS);
    float sci = g3i[0] * rsqrtf(vi + EPS);
    ws[WS_BN3 + 0] = scr; ws[WS_BN3 + 1] = be3r[0] - scr * mr;
    ws[WS_BN3 + 2] = sci; ws[WS_BN3 + 3] = be3i[0] - sci * mi;
  }
}

__global__ __launch_bounds__(256) void k_a3mom(float* __restrict__ ws) {
  __shared__ float red[256][5];
  int tid = threadIdx.x, bid = blockIdx.x;
  const float sc3r = ws[WS_BN3 + 0], sh3r = ws[WS_BN3 + 1];
  const float sc3i = ws[WS_BN3 + 2], sh3i = ws[WS_BN3 + 3];
  const float2* y3buf = (const float2*)(ws + WS_Y3BUF);
  float m0 = 0, m1 = 0, m2 = 0, m3 = 0, m4 = 0;
  for (int t = 0; t < 8; ++t) {
    float2 y = y3buf[bid * 2048 + t * 256 + tid];
    float ar = fmaxf(sc3r * y.x + sh3r, 0.f);
    float ai = fmaxf(sc3i * y.y + sh3i, 0.f);
    m0 += ar; m1 += ai; m2 += ar * ar; m3 += ai * ai; m4 += ar * ai;
  }
  red[tid][0] = m0; red[tid][1] = m1; red[tid][2] = m2; red[tid][3] = m3; red[tid][4] = m4;
  __syncthreads();
  for (int s = 128; s > 0; s >>= 1) {
    if (tid < s)
      for (int j = 0; j < 5; ++j) red[tid][j] += red[tid + s][j];
    __syncthreads();
  }
  if (tid == 0)
    for (int j = 0; j < 5; ++j) ws[WS_A3PART + bid * 5 + j] = red[0][j];
}

__global__ __launch_bounds__(32) void k_fin4(
    const float* __restrict__ w4r, const float* __restrict__ w4i,
    const float* __restrict__ b4r, const float* __restrict__ b4i,
    const float* __restrict__ g4r, const float* __restrict__ be4r,
    const float* __restrict__ g4i, const float* __restrict__ be4i,
    float* __restrict__ ws) {
  __shared__ float s5[5];
  int tid = threadIdx.x;
  if (tid < 5) {
    float s = 0;
    for (int b = 0; b < GEMM_BLOCKS; ++b) s += ws[WS_A3PART + b * 5 + tid];
    s5[tid] = s;
  }
  __syncthreads();
  if (tid == 0) {
    const float invNP = 1.f / (float)NPIX;
    float Sr = s5[0], Si = s5[1], Qr = s5[2], Qi = s5[3], X = s5[4];
    float wr = w4r[0], wi = w4i[0], br = b4r[0], bi = b4i[0];
    // y4 = w4 * a3 + b4   (complex scalar) -> analytic moments
    float sumr  = wr * Sr - wi * Si + (float)NPIX * br;
    float sumr2 = wr * wr * Qr + wi * wi * Qi - 2.f * wr * wi * X +
                  2.f * br * (wr * Sr - wi * Si) + (float)NPIX * br * br;
    float sumi  = wi * Sr + wr * Si + (float)NPIX * bi;
    float sumi2 = wi * wi * Qr + wr * wr * Qi + 2.f * wr * wi * X +
                  2.f * bi * (wi * Sr + wr * Si) + (float)NPIX * bi * bi;
    float mr = sumr * invNP, varr = sumr2 * invNP - mr * mr;
    float mi = sumi * invNP, vari = sumi2 * invNP - mi * mi;
    float sc4r = g4r[0] * rsqrtf(varr + EPS);
    float sc4i = g4i[0] * rsqrtf(vari + EPS);
    float sh4r = be4r[0] - sc4r * mr;
    float sh4i = be4i[0] - sc4i * mi;
    ws[WS_BN4 + 0] = sc4r * wr;          // enc_r = relu(ar*a3r + br_*a3i + gr)
    ws[WS_BN4 + 1] = -sc4r * wi;
    ws[WS_BN4 + 2] = sc4r * br + sh4r;
    ws[WS_BN4 + 3] = sc4i * wi;          // enc_i = relu(ai*a3r + bi_*a3i + gi)
    ws[WS_BN4 + 4] = sc4i * wr;
    ws[WS_BN4 + 5] = sc4i * bi + sh4i;
  }
}

// ------- elementwise BN3/L4/BN4 + 128-pt iFFT per (n,t) column -------
__global__ __launch_bounds__(128) void k_fft(float* __restrict__ ws) {
  __shared__ float lre[128], lim[128];
  const int bid = blockIdx.x;
  const int n = bid >> 9, t = bid & 511;
  const int j = threadIdx.x;
  const float sc3r = ws[WS_BN3 + 0], sh3r = ws[WS_BN3 + 1];
  const float sc3i = ws[WS_BN3 + 2], sh3i = ws[WS_BN3 + 3];
  const float ar = ws[WS_BN4 + 0], br = ws[WS_BN4 + 1], gr = ws[WS_BN4 + 2];
  const float ai = ws[WS_BN4 + 3], bi = ws[WS_BN4 + 4], gi = ws[WS_BN4 + 5];
  const float2* y3buf = (const float2*)(ws + WS_Y3BUF);
  float2* frames = (float2*)(ws + WS_FRAMES);
  float2 y = y3buf[n * PPI + j * TDIM + t];
  float a3r = fmaxf(sc3r * y.x + sh3r, 0.f);
  float a3i = fmaxf(sc3i * y.y + sh3i, 0.f);
  float er = fmaxf(ar * a3r + br * a3i + gr, 0.f);
  float ei = fmaxf(ai * a3r + bi * a3i + gi, 0.f);
  int rj = __brev((unsigned)j) >> 25;  // 7-bit bit reversal
  lre[rj] = er;
  lim[rj] = ei;
  __syncthreads();
  for (int len = 2; len <= 128; len <<= 1) {
    int half = len >> 1;
    int pos = j & (len - 1);
    if (pos < half) {
      int p = j + half;
      float ang = 6.28318530717958647692f * (float)pos / (float)len;  // + for iFFT
      float sn, cs;
      __sincosf(ang, &sn, &cs);
      float vr = lre[p] * cs - lim[p] * sn;
      float vi = lre[p] * sn + lim[p] * cs;
      float ur = lre[j], ui = lim[j];
      lre[j] = ur + vr; lim[j] = ui + vi;
      lre[p] = ur - vr; lim[p] = ui - vi;
    }
    __syncthreads();
  }
  const float inv = 1.f / 128.f;
  frames[(n * TDIM + t) * 128 + j] = make_float2(lre[j] * inv, lim[j] * inv);
}

// ------- overlap-add with closed-form envelope, trim n_fft/2 -------
__global__ __launch_bounds__(256) void k_ola(const float* __restrict__ ws,
                                             float* __restrict__ out) {
  int idx = blockIdx.x * 256 + threadIdx.x;
  if (idx >= N_IMG * OUTLEN) return;
  int n = idx / OUTLEN, l = idx - n * OUTLEN;
  int g = l + 64;                    // position before trimming
  int tmin = (g - 96) >> 5; if (tmin < 0) tmin = 0;   // ceil((g-127)/32)
  int tmax = g >> 5;       if (tmax > 511) tmax = 511;
  const float2* frames = (const float2*)(ws + WS_FRAMES);
  float sr = 0.f, si = 0.f;
  for (int t = tmin; t <= tmax; ++t) {
    float2 f = frames[(n * TDIM + t) * 128 + (g - t * 32)];
    sr += f.x; si += f.y;
  }
  float inv = 1.f / (float)(tmax - tmin + 1);
  out[idx * 2]     = sr * inv;
  out[idx * 2 + 1] = si * inv;
}

extern "C" void kernel_launch(void* const* d_in, const int* in_sizes, int n_in,
                              void* d_out, int out_size, void* d_ws, size_t ws_size,
                              hipStream_t stream) {
  const float2* sig  = (const float2*)d_in[0];
  const float2* dly  = (const float2*)d_in[1];
  const float2* stab = (const float2*)d_in[2];
  const int*    sid  = (const int*)d_in[3];
  // params: 4 layers x (w_r, w_i, b_r, b_i, g_r, be_r, g_i, be_i)
  const float* p0 = (const float*)0;  (void)p0;
  #define PAR(L, J) ((const float*)d_in[4 + (L) * 8 + (J)])
  float* ws  = (float*)d_ws;
  float* out = (float*)d_out;

  k_const<<<1, 1024, 0, stream>>>(dly, stab, sid,
                                  PAR(0, 0), PAR(0, 1), PAR(0, 2), PAR(0, 3), ws);
  k_sigmom<<<2048, 256, 0, stream>>>(sig, ws);
  k_fin1<<<1, 256, 0, stream>>>(PAR(0, 0), PAR(0, 1), PAR(0, 4), PAR(0, 5),
                                PAR(0, 6), PAR(0, 7), PAR(1, 0), PAR(1, 1),
                                PAR(1, 2), PAR(1, 3), ws);
  k_gemm_stats<<<GEMM_BLOCKS, 256, 0, stream>>>(sig, ws);
  k_fin2<<<1, 64, 0, stream>>>(PAR(1, 4), PAR(1, 5), PAR(1, 6), PAR(1, 7), ws);
  k_gemm_y3<<<GEMM_BLOCKS, 256, 0, stream>>>(sig, ws, PAR(2, 0), PAR(2, 1),
                                             PAR(2, 2), PAR(2, 3));
  k_fin3<<<1, 128, 0, stream>>>(PAR(2, 4), PAR(2, 5), PAR(2, 6), PAR(2, 7), ws);
  k_a3mom<<<GEMM_BLOCKS, 256, 0, stream>>>(ws);
  k_fin4<<<1, 32, 0, stream>>>(PAR(3, 0), PAR(3, 1), PAR(3, 2), PAR(3, 3),
                               PAR(3, 4), PAR(3, 5), PAR(3, 6), PAR(3, 7), ws);
  k_fft<<<N_IMG * TDIM, 128, 0, stream>>>(ws);
  k_ola<<<(N_IMG * OUTLEN + 255) / 256, 256, 0, stream>>>(ws, out);
  #undef PAR
}
